// SeqAttention_66941360275803
// MI455X (gfx1250) — matrix operand order repaired and checked by reading, not compiled, not run on hardware
//
#include <hip/hip_runtime.h>
#include <cstdint>
#include <cstddef>

// CDNA5 / gfx1250, wave32. bf16 WMMA flash-attention over a banded window.
// Pre-passes convert K -> bf16 [k][d], V -> bf16 transposed [d][k], pe -> bf16 [j][d]
// so all WMMA B-fragments in the hot loop are raw 2x global_load_b128.
// All f32<->bf16 conversions use native __bf16 casts (RTNE) so clang can pick
// hardware cvt instructions.

typedef __attribute__((ext_vector_type(16))) __bf16 v16bf;
typedef __attribute__((ext_vector_type(8)))  float  v8f;

#define BH_ 64
#define M_  1024
#define L_  1024
#define D_  64
#define KV_ (M_ + L_)

union AB16 { v16bf v; unsigned u[8]; uint4 q[2]; __bf16 h[16]; };

__device__ __forceinline__ unsigned pack2bf(float a, float b) {
  union { __bf16 h[2]; unsigned u; } r;
  r.h[0] = (__bf16)a;
  r.h[1] = (__bf16)b;
  return r.u;
}

// -------- pre-pass 1: key_pe [D][L] f32  ->  peT [L][D] bf16 --------
__global__ void pe_transpose_kernel(const float* __restrict__ pe,
                                    __bf16* __restrict__ peT) {
  int idx = blockIdx.x * blockDim.x + threadIdx.x;   // 0 .. 65535
  int j = idx >> 6, d = idx & 63;
  peT[j * D_ + d] = (__bf16)pe[d * L_ + j];
}

// -------- pre-pass 2: K f32 [b][k][d] -> Kb bf16 same layout --------
__global__ void kconv_kernel(const float* __restrict__ K,
                             unsigned* __restrict__ Kb) {
  size_t i = (size_t)blockIdx.x * blockDim.x + threadIdx.x;  // one u32 (2 bf16)
  float2 f = ((const float2*)K)[i];
  Kb[i] = pack2bf(f.x, f.y);
}

// -------- pre-pass 3: V f32 [b][k][d] -> Vt bf16 [b][d][k] (LDS tiled) --------
__global__ __launch_bounds__(256) void vtrans_kernel(const float* __restrict__ V,
                                                     unsigned* __restrict__ Vt) {
  __shared__ float tile[64][65];
  const int b  = blockIdx.y;
  const int k0 = blockIdx.x * 64;
  {
    const int kr = threadIdx.x >> 6;   // 0..3
    const int d  = threadIdx.x & 63;
    #pragma unroll
    for (int r = 0; r < 16; ++r) {
      const int k = kr * 16 + r;
      tile[k][d] = V[((size_t)b * KV_ + k0 + k) * D_ + d];
    }
  }
  __syncthreads();
  {
    const int d8 = threadIdx.x >> 5;   // 0..7
    const int c  = threadIdx.x & 31;   // u32 column within 64-k tile
    #pragma unroll
    for (int r = 0; r < 8; ++r) {
      const int d = d8 * 8 + r;
      Vt[(size_t)(b * D_ + d) * (KV_ / 2) + (k0 >> 1) + c] =
          pack2bf(tile[2 * c][d], tile[2 * c + 1][d]);
    }
  }
}

// -------- main: banded flash attention --------
// grid (M_/128, BH_), block 256 (8 waves); each wave owns 16 query rows.
__global__ __launch_bounds__(256) void seq_attn_kernel(
    const float* __restrict__ Q, const __bf16* __restrict__ Kb,
    const __bf16* __restrict__ Vt, const __bf16* __restrict__ peT,
    float* __restrict__ Out) {
  // per-wave private LDS: pos table T (bf16) + P staging for A-fragment relayout
  __shared__ __bf16 Tl[8][16][L_];   // 256 KB
  __shared__ __bf16 Pst[8][16][32];  //   8 KB

  const int wave = threadIdx.x >> 5;
  const int lane = threadIdx.x & 31;
  const int half = lane >> 4;        // lane group (0: lanes 0-15, 1: 16-31)
  const int n    = lane & 15;
  const int b    = blockIdx.y;
  const int qm0  = blockIdx.x * 128 + wave * 16;

  // fold 1/sqrt(D) and log2(e) into Q so we can use exp2
  const float SC = 0.125f * 1.44269504088896340736f;

  // ---- Q A-fragments (16x32 bf16), two K-chunks covering D=64 ----
  AB16 Qa[2];
  {
    const float* qrow = Q + ((size_t)b * M_ + qm0 + n) * D_;  // A row m = lane&15
    #pragma unroll
    for (int c = 0; c < 2; ++c) {
      const int b0 = c * 32 + half * 16;
      float4 a0 = *(const float4*)(qrow + b0);
      float4 a1 = *(const float4*)(qrow + b0 + 4);
      float4 c0 = *(const float4*)(qrow + b0 + 8);
      float4 c1 = *(const float4*)(qrow + b0 + 12);
      Qa[c].u[0] = pack2bf(a0.x * SC, a0.y * SC);
      Qa[c].u[1] = pack2bf(a0.z * SC, a0.w * SC);
      Qa[c].u[2] = pack2bf(a1.x * SC, a1.y * SC);
      Qa[c].u[3] = pack2bf(a1.z * SC, a1.w * SC);
      Qa[c].u[4] = pack2bf(c0.x * SC, c0.y * SC);
      Qa[c].u[5] = pack2bf(c0.z * SC, c0.w * SC);
      Qa[c].u[6] = pack2bf(c1.x * SC, c1.y * SC);
      Qa[c].u[7] = pack2bf(c1.z * SC, c1.w * SC);
    }
  }

  // ---- precompute pos table  T[m][j] = (Q*SC) . pe[:,j]  into LDS (bf16) ----
  for (int jt = 0; jt < 64; ++jt) {
    v8f acc = {0.f, 0.f, 0.f, 0.f, 0.f, 0.f, 0.f, 0.f};
    #pragma unroll
    for (int c = 0; c < 2; ++c) {
      AB16 Bp;  // B 32x16: lane n = pe column j, 16 consecutive d per lane
      const uint4* src =
          (const uint4*)(peT + (size_t)(jt * 16 + n) * D_ + c * 32 + half * 16);
      Bp.q[0] = src[0];
      Bp.q[1] = src[1];
      acc = __builtin_amdgcn_wmma_f32_16x16x32_bf16(
          false, Qa[c].v, false, Bp.v, (short)0, acc, false, false);
    }
    #pragma unroll
    for (int v = 0; v < 8; ++v)
      Tl[wave][v + 8 * half][jt * 16 + n] = (__bf16)acc[v];
  }

  // ---- flash loop over 33 key blocks of 32 (covers band j in [0,1024)) ----
  float mrow[8], lrow[8];
  v8f Ot[4];
  #pragma unroll
  for (int dt = 0; dt < 4; ++dt)
    Ot[dt] = (v8f){0.f, 0.f, 0.f, 0.f, 0.f, 0.f, 0.f, 0.f};
  #pragma unroll
  for (int v = 0; v < 8; ++v) { mrow[v] = -3.0e38f; lrow[v] = 0.f; }

  for (int t = 0; t < 33; ++t) {
    const int kb = qm0 + 32 * t;  // absolute key base of this block

    // scores: two 16-wide tiles, QK^T via bf16 WMMA (f32 accum)
    v8f S[2];
    #pragma unroll
    for (int nt = 0; nt < 2; ++nt) {
      v8f s = {0.f, 0.f, 0.f, 0.f, 0.f, 0.f, 0.f, 0.f};
      const int krow = min(kb + nt * 16 + n, KV_ - 1);  // clamp OOB (masked)
      const __bf16* kr = Kb + ((size_t)b * KV_ + krow) * D_;
      #pragma unroll
      for (int c = 0; c < 2; ++c) {
        AB16 Bk;  // B 32x16: lane n = key row, 16 consecutive d per lane
        const uint4* kp = (const uint4*)(kr + c * 32 + half * 16);
        Bk.q[0] = kp[0];
        Bk.q[1] = kp[1];
        s = __builtin_amdgcn_wmma_f32_16x16x32_bf16(
            false, Qa[c].v, false, Bk.v, (short)0, s, false, false);
      }
      S[nt] = s;
    }

    // skewed pos add + band mask + online softmax (row = v + 8*half)
    float corr[8];
    #pragma unroll
    for (int v = 0; v < 8; ++v) {
      const int m  = v + 8 * half;
      const int j0 = 32 * t + n - m;      // relative index of tile-0 element
      const int j1 = j0 + 16;             // tile-1
      float s0 =
          (j0 >= 0 && j0 < L_) ? S[0][v] + (float)Tl[wave][m][j0] : -3.0e38f;
      float s1 =
          (j1 >= 0 && j1 < L_) ? S[1][v] + (float)Tl[wave][m][j1] : -3.0e38f;
      float mx = fmaxf(s0, s1);
      #pragma unroll
      for (int off = 1; off < 16; off <<= 1)
        mx = fmaxf(mx, __shfl_xor(mx, off, 32));
      const float mn = fmaxf(mrow[v], mx);
      const float c0 = exp2f(mrow[v] - mn);
      const float e0 = exp2f(s0 - mn);
      const float e1 = exp2f(s1 - mn);
      float sum = e0 + e1;
      #pragma unroll
      for (int off = 1; off < 16; off <<= 1)
        sum += __shfl_xor(sum, off, 32);
      lrow[v] = lrow[v] * c0 + sum;
      mrow[v] = mn;
      corr[v] = c0;
      Pst[wave][m][n]      = (__bf16)e0;   // stage P for A-fragment relayout
      Pst[wave][m][16 + n] = (__bf16)e1;
    }
    #pragma unroll
    for (int dt = 0; dt < 4; ++dt)
      #pragma unroll
      for (int v = 0; v < 8; ++v) Ot[dt][v] *= corr[v];

    // P as A-fragment (16x32 bf16) read back from LDS in A layout
    AB16 Pa;
    {
      const unsigned* prow = (const unsigned*)&Pst[wave][n][0];  // A row = lane&15
      #pragma unroll
      for (int v = 0; v < 8; ++v) {
        const int kbase = half * 16 + v * 2;
        Pa.u[v] = prow[kbase >> 1];
      }
    }

    // O += P @ V : four 16-wide d tiles; B-fragment is 16 consecutive bf16 of Vt
    const int kvb = min(kb + half * 16, KV_ - 16);  // base clamp (clamped rows have p==0)
    #pragma unroll
    for (int dt = 0; dt < 4; ++dt) {
      AB16 Bv;  // B 32x16: lane n = d column; 16 consecutive k per lane from Vt
      const uint4* vp =
          (const uint4*)(Vt + ((size_t)b * D_ + dt * 16 + n) * KV_ + kvb);
      Bv.q[0] = vp[0];
      Bv.q[1] = vp[1];
      Ot[dt] = __builtin_amdgcn_wmma_f32_16x16x32_bf16(
          false, Pa.v, false, Bv.v, (short)0, Ot[dt], false, false);
    }
  }

  // ---- epilogue: normalize and store ----
  #pragma unroll
  for (int dt = 0; dt < 4; ++dt) {
    #pragma unroll
    for (int v = 0; v < 8; ++v) {
      const float inv = 1.0f / lrow[v];
      Out[((size_t)b * M_ + qm0 + v + 8 * half) * D_ + dt * 16 + n] =
          Ot[dt][v] * inv;
    }
  }
}

extern "C" void kernel_launch(void* const* d_in, const int* in_sizes, int n_in,
                              void* d_out, int out_size, void* d_ws, size_t ws_size,
                              hipStream_t stream) {
  const float* q  = (const float*)d_in[0];   // [64,1024,64]
  const float* k  = (const float*)d_in[1];   // [64,2048,64]
  const float* v  = (const float*)d_in[2];   // [64,2048,64]
  const float* pe = (const float*)d_in[3];   // [1,64,1024]
  float* out = (float*)d_out;                // [64,1024,64]

  // ws layout (needs ~33.7 MB):
  //   [0, 128K)            peT  bf16 [L][D]
  //   [128K, 128K+16M)     Kb   bf16 [b][k][d]
  //   [128K+16M, ...)      Vt   bf16 [b][d][k]
  __bf16* peT = (__bf16*)d_ws;
  __bf16* KbP = peT + (size_t)L_ * D_;
  __bf16* VtP = KbP + (size_t)BH_ * KV_ * D_;

  pe_transpose_kernel<<<dim3(256), dim3(256), 0, stream>>>(pe, peT);
  kconv_kernel<<<dim3((BH_ * KV_ * D_ / 2) / 256), dim3(256), 0, stream>>>(
      k, (unsigned*)KbP);
  vtrans_kernel<<<dim3(KV_ / 64, BH_), dim3(256), 0, stream>>>(v, (unsigned*)VtP);
  seq_attn_kernel<<<dim3(M_ / 128, BH_), dim3(256), 0, stream>>>(
      q, KbP, VtP, peT, out);
}
